// DeepMambaModel_18253611008303
// MI455X (gfx1250) — compile-verified, matrix operand order checked by reading, not run on hardware
//
#include <hip/hip_runtime.h>
#include <hip/hip_bf16.h>

typedef __attribute__((ext_vector_type(16))) _Float16 v16h;
typedef __attribute__((ext_vector_type(8)))  _Float16 v8h;
typedef __attribute__((ext_vector_type(8)))  float    v8f;

namespace {
constexpr int BB = 4, LL = 64, DD = 512;
constexpr int DI = 1024, NS = 16, DCV = 4, RR = 32;
constexpr int NLY = 4, SDIM = 17, ADIM = 6;
constexpr int TT = 4 * LL;            // 256 interleaved tokens per batch
constexpr int MM = BB * TT;           // 1024 GEMM rows
constexpr float LN_EPS = 1e-5f;
}

// ---------------------------------------------------------------- helpers
__device__ __forceinline__ float silu_f(float x)     { return x / (1.0f + __expf(-x)); }
__device__ __forceinline__ float softplus_f(float x) { return x > 20.0f ? x : log1pf(__expf(x)); }

// Load one 16x32 f16 WMMA operand fragment (A-layout; W rows use the same
// layout since B[k][n] == W[n][k] and both are K-contiguous).
// Per ISA 7.12.2: lanes 0-15 hold K={0..7,16..23}, lanes 16-31 K={8..15,24..31}.
__device__ __forceinline__ v16h load_frag16x32(const _Float16* __restrict__ base,
                                               int row, int ld, int k0, int lane) {
  const int kb = k0 + ((lane & 16) ? 8 : 0);
  const v8h lo = *reinterpret_cast<const v8h*>(base + (size_t)row * ld + kb);
  const v8h hi = *reinterpret_cast<const v8h*>(base + (size_t)row * ld + kb + 16);
  v16h f;
#pragma unroll
  for (int i = 0; i < 8; ++i) { f[i] = lo[i]; f[i + 8] = hi[i]; }
  return f;
}

// ---------------------------------------------------------------- GEMM (WMMA)
// C[M,N] = A[M,K] @ W[N,K]^T  (torch Linear layout).
// Block: 128 thr = 4 waves in 2x2; each wave owns a 32x32 output tile:
// 2 A frags x 2 B frags -> 4 v_wmma per k-step (each fragment feeds 2 WMMAs).
// Block tile 64x64; all problem dims divide 64 so EXEC stays all-ones.
enum { EPI_STORE = 0, EPI_BIAS_SOFTPLUS = 1, EPI_ACCUM = 2 };

template <int EPI>
__global__ void gemm_wmma_kernel(const _Float16* __restrict__ A,
                                 const _Float16* __restrict__ W,
                                 float* __restrict__ C,
                                 const float* __restrict__ bias,
                                 int M, int N, int K) {
  const int lane = threadIdx.x & 31;
  const int wave = threadIdx.x >> 5;          // 0..3
  const int m0 = blockIdx.y * 64 + (wave >> 1) * 32;
  const int n0 = blockIdx.x * 64 + (wave & 1) * 32;
  const int rA0 = m0 + (lane & 15);
  const int rA1 = rA0 + 16;
  const int rB0 = n0 + (lane & 15);
  const int rB1 = rB0 + 16;
  v8f acc00 = {}, acc01 = {}, acc10 = {}, acc11 = {};
  for (int k0 = 0; k0 < K; k0 += 32) {
    v16h a0 = load_frag16x32(A, rA0, K, k0, lane);
    v16h a1 = load_frag16x32(A, rA1, K, k0, lane);
    v16h b0 = load_frag16x32(W, rB0, K, k0, lane);
    v16h b1 = load_frag16x32(W, rB1, K, k0, lane);
    acc00 = __builtin_amdgcn_wmma_f32_16x16x32_f16(false, a0, false, b0, (short)0, acc00, false, false);
    acc01 = __builtin_amdgcn_wmma_f32_16x16x32_f16(false, a0, false, b1, (short)0, acc01, false, false);
    acc10 = __builtin_amdgcn_wmma_f32_16x16x32_f16(false, a1, false, b0, (short)0, acc10, false, false);
    acc11 = __builtin_amdgcn_wmma_f32_16x16x32_f16(false, a1, false, b1, (short)0, acc11, false, false);
  }
  // C/D layout: VGPR r -> row base+r (lanes 0-15) / base+8+r (lanes 16-31), col = lane&15
  const int rbase = ((lane & 16) ? 8 : 0);
  const int col = lane & 15;
#pragma unroll
  for (int r = 0; r < 8; ++r) {
    const int row0 = m0 + rbase + r;
    const int row1 = row0 + 16;
    const int c0 = n0 + col, c1 = n0 + 16 + col;
    float v00 = acc00[r], v01 = acc01[r], v10 = acc10[r], v11 = acc11[r];
    if (EPI == EPI_BIAS_SOFTPLUS) {
      v00 = softplus_f(v00 + bias[c0]);
      v01 = softplus_f(v01 + bias[c1]);
      v10 = softplus_f(v10 + bias[c0]);
      v11 = softplus_f(v11 + bias[c1]);
    }
    if (EPI == EPI_ACCUM) {
      C[(size_t)row0 * N + c0] += v00;
      C[(size_t)row0 * N + c1] += v01;
      C[(size_t)row1 * N + c0] += v10;
      C[(size_t)row1 * N + c1] += v11;
    } else {
      C[(size_t)row0 * N + c0] = v00;
      C[(size_t)row0 * N + c1] = v01;
      C[(size_t)row1 * N + c0] = v10;
      C[(size_t)row1 * N + c1] = v11;
    }
  }
}

// ---------------------------------------------------------------- embeddings
__global__ void embed_kernel(const float* __restrict__ states, const float* __restrict__ actions,
                             const float* __restrict__ rtg, const float* __restrict__ ctg,
                             const int* __restrict__ timesteps,
                             const float* __restrict__ es_w, const float* __restrict__ es_b,
                             const float* __restrict__ ea_w, const float* __restrict__ ea_b,
                             const float* __restrict__ er_w, const float* __restrict__ er_b,
                             const float* __restrict__ ec_w, const float* __restrict__ ec_b,
                             const float* __restrict__ et_w,
                             float* __restrict__ x) {
  const int idx = blockIdx.x * blockDim.x + threadIdx.x;   // over MM*DD
  if (idx >= MM * DD) return;
  const int d = idx % DD;
  const int tok = idx / DD;
  const int b = tok / TT;
  const int t = tok % TT;
  const int l = t >> 2;
  const int j = t & 3;                 // 0=return 1=constraint 2=state 3=action
  const int bl = b * LL + l;
  const float te = et_w[(size_t)timesteps[bl] * DD + d];
  float v;
  if (j == 0) {
    v = rtg[bl] * er_w[d] + er_b[d];
  } else if (j == 1) {
    v = ctg[bl] * ec_w[d] + ec_b[d];
  } else if (j == 2) {
    v = es_b[d];
    for (int s = 0; s < SDIM; ++s) v += states[bl * SDIM + s] * es_w[d * SDIM + s];
  } else {
    v = ea_b[d];
    for (int a = 0; a < ADIM; ++a) v += actions[bl * ADIM + a] * ea_w[d * ADIM + a];
  }
  x[idx] = v + te;
}

// ---------------------------------------------------------------- layernorm
__global__ void layernorm_kernel(const float* __restrict__ x,
                                 const float* __restrict__ w, const float* __restrict__ b,
                                 _Float16* __restrict__ out16, float* __restrict__ out32) {
  const int row = blockIdx.x;
  const int tx = threadIdx.x;
  const float* xr = x + (size_t)row * DD;
  __shared__ float red[256];
  float s = 0.0f;
  for (int d = tx; d < DD; d += 256) s += xr[d];
  red[tx] = s; __syncthreads();
  for (int off = 128; off > 0; off >>= 1) { if (tx < off) red[tx] += red[tx + off]; __syncthreads(); }
  const float mu = red[0] * (1.0f / DD);
  __syncthreads();
  float v = 0.0f;
  for (int d = tx; d < DD; d += 256) { float t = xr[d] - mu; v += t * t; }
  red[tx] = v; __syncthreads();
  for (int off = 128; off > 0; off >>= 1) { if (tx < off) red[tx] += red[tx + off]; __syncthreads(); }
  const float rstd = rsqrtf(red[0] * (1.0f / DD) + LN_EPS);
  for (int d = tx; d < DD; d += 256) {
    const float y = (xr[d] - mu) * rstd * w[d] + b[d];
    if (out16) out16[(size_t)row * DD + d] = (_Float16)y;
    if (out32) out32[(size_t)row * DD + d] = y;
  }
}

// ---------------------------------------------------------------- converts
__global__ void cvt_f16_kernel(const float* __restrict__ in, _Float16* __restrict__ out, int n) {
  for (int i = blockIdx.x * blockDim.x + threadIdx.x; i < n; i += gridDim.x * blockDim.x)
    out[i] = (_Float16)in[i];
}

__global__ void extract_dt_kernel(const float* __restrict__ dbc, _Float16* __restrict__ dt16) {
  const int idx = blockIdx.x * blockDim.x + threadIdx.x;   // over MM*RR
  if (idx >= MM * RR) return;
  const int r = idx % RR;
  const int bt = idx / RR;
  dt16[idx] = (_Float16)dbc[bt * (RR + 2 * NS) + r];
}

// ---------------------------------------------------------------- conv + silu
__global__ void conv_silu_kernel(const float* __restrict__ xz,
                                 const float* __restrict__ conv_w, const float* __restrict__ conv_b,
                                 float* __restrict__ xc, _Float16* __restrict__ xc16) {
  const int idx = blockIdx.x * blockDim.x + threadIdx.x;   // over MM*DI
  if (idx >= MM * DI) return;
  const int d = idx % DI;
  const int bt = idx / DI;
  const int t = bt % TT;
  const int b = bt / TT;
  float acc = conv_b[d];
#pragma unroll
  for (int k = 0; k < DCV; ++k) {
    const int tt = t - (DCV - 1) + k;
    if (tt >= 0) acc += xz[((size_t)(b * TT + tt)) * (2 * DI) + d] * conv_w[d * DCV + k];
  }
  const float s = silu_f(acc);
  xc[idx] = s;
  xc16[idx] = (_Float16)s;
}

// ---------------------------------------------------------------- selective scan + gate
__global__ void scan_gate_kernel(const float* __restrict__ xc, const float* __restrict__ delta,
                                 const float* __restrict__ dbc, const float* __restrict__ A_log,
                                 const float* __restrict__ Dp, const float* __restrict__ xz,
                                 _Float16* __restrict__ g16) {
  const int d = blockIdx.x * 256 + threadIdx.x;   // channel within DI
  const int b = blockIdx.y;
  float An[NS], h[NS];
#pragma unroll
  for (int n = 0; n < NS; ++n) { An[n] = -__expf(A_log[d * NS + n]); h[n] = 0.0f; }
  const float Dv = Dp[d];
  __shared__ float sBC[2 * NS];
  for (int t = 0; t < TT; ++t) {
    const int bt = b * TT + t;
    if (threadIdx.x < 2 * NS) sBC[threadIdx.x] = dbc[bt * (RR + 2 * NS) + RR + threadIdx.x];
    __syncthreads();
    const float u  = xc[(size_t)bt * DI + d];
    const float dl = delta[(size_t)bt * DI + d];
    float y = 0.0f;
#pragma unroll
    for (int n = 0; n < NS; ++n) {
      const float dA = __expf(dl * An[n]);
      h[n] = dA * h[n] + dl * sBC[n] * u;
      y += h[n] * sBC[NS + n];
    }
    y += u * Dv;
    const float z = xz[(size_t)bt * (2 * DI) + DI + d];
    g16[(size_t)bt * DI + d] = (_Float16)(y * silu_f(z));
    __syncthreads();
  }
}

// ---------------------------------------------------------------- heads
__global__ void head_kernel(const float* __restrict__ xf,
                            const float* __restrict__ ps_w, const float* __restrict__ ps_b,
                            const float* __restrict__ pa_w, const float* __restrict__ pa_b,
                            float* __restrict__ out) {
  const int idx = blockIdx.x * blockDim.x + threadIdx.x;
  const int nState = BB * LL * SDIM;
  const int total  = nState + BB * LL * ADIM;
  if (idx >= total) return;
  if (idx < nState) {
    const int s = idx % SDIM; const int bl = idx / SDIM;
    const int l = bl % LL;    const int b = bl / LL;
    const float* tok = xf + ((size_t)(b * TT + 4 * l + 3)) * DD;   // action tokens -> state preds
    float acc = ps_b[s];
    for (int d = 0; d < DD; ++d) acc += tok[d] * ps_w[s * DD + d];
    out[bl * SDIM + s] = acc;
  } else {
    const int j = idx - nState;
    const int a = j % ADIM; const int bl = j / ADIM;
    const int l = bl % LL;  const int b = bl / LL;
    const float* tok = xf + ((size_t)(b * TT + 4 * l + 2)) * DD;   // state tokens -> action preds
    float acc = pa_b[a];
    for (int d = 0; d < DD; ++d) acc += tok[d] * pa_w[a * DD + d];
    out[nState + bl * ADIM + a] = acc;
  }
}

// ---------------------------------------------------------------- launch
static inline void* wsa(char*& p, size_t bytes) {
  void* r = (void*)p;
  p += (bytes + 255) & ~((size_t)255);
  return r;
}

extern "C" void kernel_launch(void* const* d_in, const int* in_sizes, int n_in,
                              void* d_out, int out_size, void* d_ws, size_t ws_size,
                              hipStream_t stream) {
  (void)in_sizes; (void)n_in; (void)out_size; (void)ws_size;
  const float* states  = (const float*)d_in[0];
  const float* actions = (const float*)d_in[1];
  const float* rtg     = (const float*)d_in[2];
  const float* ctg     = (const float*)d_in[3];
  const int*   tsteps  = (const int*)  d_in[4];
  const float* es_w = (const float*)d_in[5];  const float* es_b = (const float*)d_in[6];
  const float* ea_w = (const float*)d_in[7];  const float* ea_b = (const float*)d_in[8];
  const float* er_w = (const float*)d_in[9];  const float* er_b = (const float*)d_in[10];
  const float* ec_w = (const float*)d_in[11]; const float* ec_b = (const float*)d_in[12];
  const float* et_w = (const float*)d_in[13];
  const float* ln_w = (const float*)d_in[14]; const float* ln_b = (const float*)d_in[15];
  const float* in_w = (const float*)d_in[16];
  const float* conv_w = (const float*)d_in[17]; const float* conv_b = (const float*)d_in[18];
  const float* xp_w = (const float*)d_in[19];
  const float* dtp_w = (const float*)d_in[20]; const float* dtp_b = (const float*)d_in[21];
  const float* A_log = (const float*)d_in[22]; const float* Dp = (const float*)d_in[23];
  const float* out_w = (const float*)d_in[24];
  const float* fn_w = (const float*)d_in[25]; const float* fn_b = (const float*)d_in[26];
  const float* ps_w = (const float*)d_in[27]; const float* ps_b = (const float*)d_in[28];
  const float* pa_w = (const float*)d_in[29]; const float* pa_b = (const float*)d_in[30];

  char* p = (char*)d_ws;
  float*    x     = (float*)   wsa(p, (size_t)MM * DD * 4);
  _Float16* xn16  = (_Float16*)wsa(p, (size_t)MM * DD * 2);
  float*    xz    = (float*)   wsa(p, (size_t)MM * 2 * DI * 4);
  float*    xc    = (float*)   wsa(p, (size_t)MM * DI * 4);
  _Float16* xc16  = (_Float16*)wsa(p, (size_t)MM * DI * 2);
  float*    dbc   = (float*)   wsa(p, (size_t)MM * (RR + 2 * NS) * 4);
  _Float16* dt16  = (_Float16*)wsa(p, (size_t)MM * RR * 2);
  float*    delta = (float*)   wsa(p, (size_t)MM * DI * 4);
  _Float16* g16   = (_Float16*)wsa(p, (size_t)MM * DI * 2);
  _Float16* w16   = (_Float16*)wsa(p, (size_t)2 * DI * DD * 2);  // largest weight (in_w layer)
  float*    xf    = (float*)   wsa(p, (size_t)MM * DD * 4);

  // 1. embeddings + (r,c,s,a) interleave
  embed_kernel<<<(MM * DD + 255) / 256, 256, 0, stream>>>(
      states, actions, rtg, ctg, tsteps, es_w, es_b, ea_w, ea_b,
      er_w, er_b, ec_w, ec_b, et_w, x);

  for (int i = 0; i < NLY; ++i) {
    // layernorm -> f16 activations
    layernorm_kernel<<<MM, 256, 0, stream>>>(x, ln_w + i * DD, ln_b + i * DD, xn16, nullptr);

    // in_proj: (1024x512) @ (2048x512)^T -> xz (1024x2048)
    {
      const int n = 2 * DI * DD;
      cvt_f16_kernel<<<2048, 256, 0, stream>>>(in_w + (size_t)i * n, w16, n);
      gemm_wmma_kernel<EPI_STORE><<<dim3((2 * DI) / 64, MM / 64), 128, 0, stream>>>(
          xn16, w16, xz, nullptr, MM, 2 * DI, DD);
    }

    // causal depthwise conv + silu
    conv_silu_kernel<<<(MM * DI) / 256, 256, 0, stream>>>(
        xz, conv_w + i * DI * DCV, conv_b + i * DI, xc, xc16);

    // x_proj: (1024x1024) @ (64x1024)^T -> dbc (1024x64)
    {
      const int n = (RR + 2 * NS) * DI;
      cvt_f16_kernel<<<256, 256, 0, stream>>>(xp_w + (size_t)i * n, w16, n);
      gemm_wmma_kernel<EPI_STORE><<<dim3(1, MM / 64), 128, 0, stream>>>(
          xc16, w16, dbc, nullptr, MM, RR + 2 * NS, DI);
    }

    // dt -> f16, then dt_proj with fused bias+softplus -> delta
    extract_dt_kernel<<<(MM * RR) / 256, 256, 0, stream>>>(dbc, dt16);
    {
      const int n = DI * RR;
      cvt_f16_kernel<<<128, 256, 0, stream>>>(dtp_w + (size_t)i * n, w16, n);
      gemm_wmma_kernel<EPI_BIAS_SOFTPLUS><<<dim3(DI / 64, MM / 64), 128, 0, stream>>>(
          dt16, w16, delta, dtp_b + i * DI, MM, DI, RR);
    }

    // selective scan + silu(z) gate -> g16
    scan_gate_kernel<<<dim3(DI / 256, BB), 256, 0, stream>>>(
        xc, delta, dbc, A_log + i * DI * NS, Dp + i * DI, xz, g16);

    // out_proj: (1024x1024) @ (512x1024)^T accumulated into residual x
    {
      const int n = DD * DI;
      cvt_f16_kernel<<<2048, 256, 0, stream>>>(out_w + (size_t)i * n, w16, n);
      gemm_wmma_kernel<EPI_ACCUM><<<dim3(DD / 64, MM / 64), 128, 0, stream>>>(
          g16, w16, x, nullptr, MM, DD, DI);
    }
  }

  // final layernorm (f32) + prediction heads
  layernorm_kernel<<<MM, 256, 0, stream>>>(x, fn_w, fn_b, nullptr, xf);
  head_kernel<<<(BB * LL * (SDIM + ADIM) + 255) / 256, 256, 0, stream>>>(
      xf, ps_w, ps_b, pa_w, pa_b, (float*)d_out);
}
